// MyGNN_66348654789163
// MI455X (gfx1250) — matrix operand order repaired and checked by reading, not compiled, not run on hardware
//
#include <hip/hip_runtime.h>
#include <hip/hip_bf16.h>

// ---- problem constants (match reference) ----
#define N_NODES    100000
#define N_EDGES    400000
#define NUM_GRAPHS 512
#define DIM_H      128
#define DIM_FC     256
#define DIM_OUT    12

typedef __attribute__((ext_vector_type(16))) __bf16 v16bf;
typedef __attribute__((ext_vector_type(8)))  float  v8f;

// ---------------------------------------------------------------------------
// WMMA GEMM:  C[M,N] = relu?( A[M,K=128] @ W[K,N] + bias )
// A bf16 row-major [m][k]; Wt bf16 pre-transposed [n][k].  N is a template
// parameter (128 or 256) so the per-wave tile count is compile-time constant:
// no divergent guards, accumulators stay pinned in VGPRs, EXEC is all-ones
// around every WMMA.  Block = 256 threads (8 waves), 16-row M tile per block;
// wave w owns N-tiles {w, w+8}.  K staged in two 64-chunks in LDS.
// ---------------------------------------------------------------------------
template <int N, int DO_RELU>
__global__ __launch_bounds__(256) void gemm_wmma_k128(
    const __bf16* __restrict__ A, const __bf16* __restrict__ Wt,
    const float* __restrict__ bias, float* __restrict__ C, int M) {
  constexpr int K = 128;
  constexpr int NT = N / 128;          // N-tiles per wave (1 or 2)
  __shared__ __bf16 sA[16 * K];        // 4 KB
  __shared__ __bf16 sWt[N * 64];       // 16/32 KB : W^T K-chunk [n][k]

  const int tid  = threadIdx.x;
  const int row0 = blockIdx.x << 4;

  // stage A tile as 32-bit copies (2 bf16 each), coalesced
  {
    const unsigned int* Au  = (const unsigned int*)A;   // 64 uints per row
    unsigned int*       sAu = (unsigned int*)sA;
    for (int i = tid; i < 16 * 64; i += 256) {
      int r = i >> 6, kw = i & 63;
      int row = row0 + r;
      sAu[i] = (row < M) ? Au[(size_t)row * 64 + kw] : 0u;
    }
  }

  const int wave = tid >> 5;
  const int lane = tid & 31;
  const int half = lane >> 4;
  const int l16  = lane & 15;

  v8f zero = {0.f, 0.f, 0.f, 0.f, 0.f, 0.f, 0.f, 0.f};
  v8f acc[NT];
#pragma unroll
  for (int ti = 0; ti < NT; ++ti) acc[ti] = zero;

#pragma unroll
  for (int kc = 0; kc < K; kc += 64) {
    __syncthreads();  // protects sWt reuse (and covers sA staging on iter 0)
    // stage W^T chunk (straight 32-bit copy)
    {
      const unsigned int* Wu  = (const unsigned int*)Wt;  // 64 uints per n-row
      unsigned int*       sWu = (unsigned int*)sWt;
      const int kw0 = kc >> 1;
      for (int i = tid; i < (N << 5); i += 256) {
        int n = i >> 5, kw = i & 31;
        sWu[i] = Wu[(size_t)n * 64 + kw0 + kw];
      }
    }
    __syncthreads();

#pragma unroll
    for (int ti = 0; ti < NT; ++ti) {
      const int ncol = ((ti * 8 + wave) << 4) + l16;
      const __bf16* bbase = &sWt[ncol * 64];
#pragma unroll
      for (int kk = 0; kk < 64; kk += 32) {
        v16bf af, bfv;
        // A 16x32 bf16 layout (ISA 7.12.2): lane half selects K-interleave
        // low 8 elems: k = 8*half + e ; high 8 elems: k = 16 + 8*half + e
        const __bf16* ap = &sA[l16 * K + kc + kk + (half << 3)];
#pragma unroll
        for (int e = 0; e < 8; ++e) af[e] = ap[e];
#pragma unroll
        for (int e = 0; e < 8; ++e) af[e + 8] = ap[e + 16];
        // B 32x16 layout: lane = n, same K interleave as A
        const __bf16* bp = bbase + kk + (half << 3);
#pragma unroll
        for (int e = 0; e < 8; ++e) bfv[e] = bp[e];
#pragma unroll
        for (int e = 0; e < 8; ++e) bfv[e + 8] = bp[e + 16];
        acc[ti] = __builtin_amdgcn_wmma_f32_16x16x32_bf16(
            false, af, false, bfv, (short)0, acc[ti], false, false);
      }
    }
  }

  // epilogue: C/D layout — VGPR r holds M = r + 8*half, N = lane&15
#pragma unroll
  for (int ti = 0; ti < NT; ++ti) {
    const int ncol = ((ti * 8 + wave) << 4) + l16;
    float bv = bias[ncol];
#pragma unroll
    for (int r = 0; r < 8; ++r) {
      int row = row0 + r + (half << 3);
      if (row < M) {
        float v = acc[ti][r] + bv;
        if (DO_RELU) v = fmaxf(v, 0.0f);
        C[(size_t)row * N + ncol] = v;
      }
    }
  }
}

// ---- converters -------------------------------------------------------------
__global__ void cvt_f32_bf16(const float* __restrict__ in, __bf16* __restrict__ out,
                             int n) {
  int i = blockIdx.x * blockDim.x + threadIdx.x;
  if (i < n) out[i] = (__bf16)in[i];
}
// W [k][n] f32  ->  Wt [n][k] bf16
__global__ void cvt_w_transpose(const float* __restrict__ W, __bf16* __restrict__ Wt,
                                int K, int N) {
  int i = blockIdx.x * blockDim.x + threadIdx.x;
  if (i >= K * N) return;
  int k = i / N, n = i % N;          // coalesced global read of W
  Wt[(size_t)n * K + k] = (__bf16)W[i];
}

// ---- degree / inverse degree ------------------------------------------------
__global__ void deg_kernel(const int* __restrict__ dst, float* __restrict__ deg) {
  int e = blockIdx.x * blockDim.x + threadIdx.x;
  if (e < N_EDGES) atomicAdd(&deg[dst[e]], 1.0f);
}
__global__ void invdeg_kernel(float* __restrict__ deg) {
  int n = blockIdx.x * blockDim.x + threadIdx.x;
  if (n < N_NODES) deg[n] = 1.0f / fmaxf(deg[n], 1.0f);
}

// ---- edge scatter: agg[dst] += h[src]  (wave covers one edge's 128 dims) ----
__global__ void scatter_add(const float* __restrict__ h, const int* __restrict__ src,
                            const int* __restrict__ dst, float* __restrict__ agg) {
  int t = blockIdx.x * blockDim.x + threadIdx.x;   // N_EDGES*32 work items
  if (t >= N_EDGES * 32) return;
  int e = t >> 5;
  int c = (t & 31) << 2;
  int s = src[e], d = dst[e];
  const float4 v = *(const float4*)&h[(size_t)s * DIM_H + c];
  float* o = &agg[(size_t)d * DIM_H + c];
  atomicAdd(o + 0, v.x);
  atomicAdd(o + 1, v.y);
  atomicAdd(o + 2, v.z);
  atomicAdd(o + 3, v.w);
}

// ---- fused act = bf16(tanh(relu(agg * inv_deg))) ----------------------------
__global__ void finish_act(const float* __restrict__ agg,
                           const float* __restrict__ inv_deg,
                           __bf16* __restrict__ act) {
  int t = blockIdx.x * blockDim.x + threadIdx.x;   // N_NODES*128
  if (t >= N_NODES * DIM_H) return;
  float v = agg[t] * inv_deg[t >> 7];
  v = fmaxf(v, 0.0f);
  act[t] = (__bf16)tanhf(v);
}

// ---- global mean pool (reads bf16 activations, f32 atomic accumulate) -------
__global__ void pool_sum(const __bf16* __restrict__ act, const int* __restrict__ batch,
                         float* __restrict__ pooled, float* __restrict__ counts) {
  int t = blockIdx.x * blockDim.x + threadIdx.x;   // N_NODES*32
  if (t >= N_NODES * 32) return;
  int n = t >> 5;
  int c = (t & 31) << 2;
  int g = batch[n];
  const __bf16* row = &act[(size_t)n * DIM_H + c];
  float* o = &pooled[(size_t)g * DIM_H + c];
  atomicAdd(o + 0, (float)row[0]);
  atomicAdd(o + 1, (float)row[1]);
  atomicAdd(o + 2, (float)row[2]);
  atomicAdd(o + 3, (float)row[3]);
  if (c == 0) atomicAdd(&counts[g], 1.0f);
}
__global__ void pool_div(float* __restrict__ pooled, const float* __restrict__ counts) {
  int i = blockIdx.x * blockDim.x + threadIdx.x;   // NUM_GRAPHS*128
  if (i < NUM_GRAPHS * DIM_H) pooled[i] /= fmaxf(counts[i >> 7], 1.0f);
}

// ---- final regressor (N=12, not WMMA-shaped; trivial FLOPs) -----------------
__global__ void reg_kernel(const float* __restrict__ g, const float* __restrict__ Wr,
                           const float* __restrict__ br, float* __restrict__ out) {
  int t = blockIdx.x * blockDim.x + threadIdx.x;   // NUM_GRAPHS*DIM_OUT
  if (t >= NUM_GRAPHS * DIM_OUT) return;
  int gi = t / DIM_OUT, o = t % DIM_OUT;
  float s = br[o];
  const float* row = &g[(size_t)gi * DIM_FC];
  for (int k = 0; k < DIM_FC; ++k) s = fmaf(row[k], Wr[(size_t)k * DIM_OUT + o], s);
  out[t] = s;
}

extern "C" void kernel_launch(void* const* d_in, const int* in_sizes, int n_in,
                              void* d_out, int out_size, void* d_ws, size_t ws_size,
                              hipStream_t stream) {
  (void)in_sizes; (void)n_in; (void)out_size; (void)ws_size;
  const float* x     = (const float*)d_in[0];
  const int*   edge  = (const int*)d_in[1];
  const int*   batch = (const int*)d_in[2];
  const float* Wl[4] = {(const float*)d_in[3], (const float*)d_in[5],
                        (const float*)d_in[7], (const float*)d_in[9]};
  const float* bl[4] = {(const float*)d_in[4], (const float*)d_in[6],
                        (const float*)d_in[8], (const float*)d_in[10]};
  const float* Wfc  = (const float*)d_in[11];
  const float* bfc  = (const float*)d_in[12];
  const float* Wreg = (const float*)d_in[13];
  const float* breg = (const float*)d_in[14];
  float* out = (float*)d_out;

  const int* src = edge;
  const int* dst = edge + N_EDGES;

  // carve workspace
  size_t off = 0;
  auto carve = [&](size_t bytes) -> void* {
    void* p = (char*)d_ws + off;
    off += (bytes + 255) & ~(size_t)255;
    return p;
  };
  const size_t featF32 = sizeof(float)  * (size_t)N_NODES * DIM_H;  // 51.2 MB
  const size_t featBF  = sizeof(__bf16) * (size_t)N_NODES * DIM_H;  // 25.6 MB
  float*  H        = (float*)carve(featF32);
  float*  AGG      = (float*)carve(featF32);
  __bf16* ACT      = (__bf16*)carve(featBF);
  float*  invd     = (float*)carve(sizeof(float) * N_NODES);
  float*  pooled   = (float*)carve(sizeof(float) * NUM_GRAPHS * DIM_H);
  __bf16* pooledbf = (__bf16*)carve(sizeof(__bf16) * NUM_GRAPHS * DIM_H);
  float*  counts   = (float*)carve(sizeof(float) * NUM_GRAPHS);
  float*  gbuf     = (float*)carve(sizeof(float) * NUM_GRAPHS * DIM_FC);
  __bf16* Wtbuf    = (__bf16*)carve(sizeof(__bf16) * 128 * 256);

  const int B = 256;
  const int gemmBlocks = N_NODES / 16;                 // 6250
  const int edgeBlocks = (N_EDGES * 32 + B - 1) / B;   // 50000
  const int featBlocks = (N_NODES * DIM_H + B - 1) / B;
  const int poolBlocks = (N_NODES * 32 + B - 1) / B;

  // degree -> inv_deg
  hipMemsetAsync(invd, 0, sizeof(float) * N_NODES, stream);
  deg_kernel<<<(N_EDGES + B - 1) / B, B, 0, stream>>>(dst, invd);
  invdeg_kernel<<<(N_NODES + B - 1) / B, B, 0, stream>>>(invd);

  // input features -> bf16
  cvt_f32_bf16<<<featBlocks, B, 0, stream>>>(x, ACT, N_NODES * DIM_H);

  // 4 message-passing layers (ACT -> H -> AGG -> ACT, stream-ordered reuse)
  for (int l = 0; l < 4; ++l) {
    cvt_w_transpose<<<(128 * 128 + B - 1) / B, B, 0, stream>>>(Wl[l], Wtbuf, 128, 128);
    gemm_wmma_k128<128, 0><<<gemmBlocks, B, 0, stream>>>(ACT, Wtbuf, bl[l], H,
                                                         N_NODES);
    hipMemsetAsync(AGG, 0, featF32, stream);
    scatter_add<<<edgeBlocks, B, 0, stream>>>(H, src, dst, AGG);
    finish_act<<<featBlocks, B, 0, stream>>>(AGG, invd, ACT);
  }

  // global mean pool
  hipMemsetAsync(pooled, 0, sizeof(float) * NUM_GRAPHS * DIM_H, stream);
  hipMemsetAsync(counts, 0, sizeof(float) * NUM_GRAPHS, stream);
  pool_sum<<<poolBlocks, B, 0, stream>>>(ACT, batch, pooled, counts);
  pool_div<<<(NUM_GRAPHS * DIM_H + B - 1) / B, B, 0, stream>>>(pooled, counts);
  cvt_f32_bf16<<<(NUM_GRAPHS * DIM_H + B - 1) / B, B, 0, stream>>>(
      pooled, pooledbf, NUM_GRAPHS * DIM_H);

  // FC (relu) + regressor
  cvt_w_transpose<<<(128 * 256 + B - 1) / B, B, 0, stream>>>(Wfc, Wtbuf, 128, 256);
  gemm_wmma_k128<256, 1><<<NUM_GRAPHS / 16, B, 0, stream>>>(pooledbf, Wtbuf, bfc,
                                                            gbuf, NUM_GRAPHS);
  reg_kernel<<<(NUM_GRAPHS * DIM_OUT + B - 1) / B, B, 0, stream>>>(gbuf, Wreg,
                                                                   breg, out);
}
